// PolyNet_90297392431134
// MI455X (gfx1250) — matrix operand order repaired and checked by reading, not compiled, not run on hardware
//
#include <hip/hip_runtime.h>
#include <hip/hip_bf16.h>

typedef __attribute__((ext_vector_type(16))) _Float16 v16h;
typedef __attribute__((ext_vector_type(8)))  _Float16 v8h;
typedef __attribute__((ext_vector_type(8)))  float    v8f;

namespace {
constexpr int kB   = 8192;
constexpr int kDIn = 64;
constexpr int kNF  = 128;
constexpr int kBM  = 32;     // rows per workgroup
constexpr int ZB_S = 104;    // [bias|x(64)] padded; 208B rows (16B aligned)
constexpr int OT_S = 136;    // 128 padded; 272B rows
constexpr int OB_S = 304;    // 288 padded (layer-3 o-part = 9 chunks); 608B rows
constexpr int HPB  = 512;    // halves per packed 32x16 B block

constexpr int L1_BLKS  = 3 * 8;    // K=65  -> 3 chunks of 32
constexpr int L2O_BLKS = 5 * 8;    // K=129 -> 5 chunks
constexpr int L3O_BLKS = 9 * 8;    // K=257 -> 9 chunks
constexpr int OUT_BLKS = 512 * 8;  // K=16384 outer-product segment

constexpr size_t oWf1 = 0;
constexpr size_t oWg1 = oWf1 + (size_t)L1_BLKS * HPB;
constexpr size_t oWf2 = oWg1 + (size_t)L1_BLKS * HPB;
constexpr size_t oWg2 = oWf2 + (size_t)(L2O_BLKS + OUT_BLKS) * HPB;
constexpr size_t oWf3 = oWg2 + (size_t)(L2O_BLKS + OUT_BLKS) * HPB;
constexpr size_t oWg3 = oWf3 + (size_t)(L3O_BLKS + OUT_BLKS) * HPB;
constexpr size_t kWsHalves = oWg3 + (size_t)(L3O_BLKS + OUT_BLKS) * HPB;
} // namespace

// ---------------------------------------------------------------------------
// Weight pack: fp32 row-major (K x 128) -> f16 WMMA B-fragments.
// Block (chunk c, ntile t): 32 lanes x 16 halves, flat [lane*16 + h].
// B 32x16 layout (ISA 7.12.2): lane L -> n = L&15, hi = L>>4;
//   halves 0..7 -> K = hi*8 + h; halves 8..15 -> K = 16 + hi*8 + (h-8).
// K rows >= kvalid are zero-padded (covers 65->96, 129->160, 257->288).
// ---------------------------------------------------------------------------
__global__ __launch_bounds__(256) void pack_w_kernel(
    const float* __restrict__ W, _Float16* __restrict__ dst,
    int rowoff, int kvalid, int nblocks)
{
  int total = nblocks << 9;
  int f = blockIdx.x * 256 + threadIdx.x;
  if (f >= total) return;
  int blk = f >> 9, within = f & 511;
  int lane = within >> 4, h = within & 15;
  int c = blk >> 3, t = blk & 7;
  int K = ((lane >> 4) << 3) + (h & 7) + ((h >> 3) << 4);
  int n = (t << 4) + (lane & 15);
  int k = (c << 5) + K;
  float v = (k < kvalid) ? W[(size_t)(rowoff + k) * kNF + n] : 0.0f;
  dst[f] = (_Float16)v;
}

// ---------------------------------------------------------------------------
// A-fragment gen (16x32 f16 A layout, ISA 7.12.2):
//   lane L (m = L&15), hi = L>>4:
//     halves 0..7  = K = hi*8 + {0..7}      -> src[off    .. off+7]
//     halves 8..15 = K = 16 + hi*8 + {0..7} -> src[off+16 .. off+23]
// For the outer-product segment every element is scaled by s = o1[m, i].
// ---------------------------------------------------------------------------
__device__ __forceinline__ v8f vzero()
{
  v8f z;
#pragma unroll
  for (int r = 0; r < 8; ++r) z[r] = 0.0f;
  return z;
}

template <bool SCALED>
__device__ __forceinline__ v16h make_a(const _Float16* __restrict__ row, int off, _Float16 s)
{
  v8h lo  = *(const v8h*)(row + off);
  v8h hi8 = *(const v8h*)(row + off + 16);
  v16h a;
#pragma unroll
  for (int i = 0; i < 8; ++i) {
    if (SCALED) { a[i] = (_Float16)(lo[i] * s); a[i + 8] = (_Float16)(hi8[i] * s); }
    else        { a[i] = lo[i];                 a[i + 8] = hi8[i]; }
  }
  return a;
}

__device__ __forceinline__ v16h load_b(const _Float16* __restrict__ w, int blk, int lane)
{
  return *(const v16h*)(w + ((size_t)blk << 9) + (lane << 4));
}

#define WMMA_F16(A, Bm, C) \
  __builtin_amdgcn_wmma_f32_16x16x32_f16(false, (A), false, (Bm), (short)0, (C), false, false)

// One K-segment of the fused GEMM. Each wave owns 2 row-tiles x 1 n-tile for
// both f and g, so every B fragment is reused by two WMMAs.
template <bool OUTER>
__device__ __forceinline__ void gemm_seg(
    const _Float16* __restrict__ src0,   // LDS row base, A rows m = ln
    const _Float16* __restrict__ src1,   // LDS row base, A rows m = 16 + ln
    const _Float16* __restrict__ scl0,   // o1 row (scale) for tile 0 when OUTER
    const _Float16* __restrict__ scl1,   // o1 row (scale) for tile 1 when OUTER
    const _Float16* __restrict__ wf, const _Float16* __restrict__ wg,
    int nch, int t0, int lane, int hi,
    v8f& f0, v8f& f1, v8f& g0, v8f& g1)
{
  for (int c = 0; c < nch; ++c) {
    int off;
    _Float16 s0 = (_Float16)1.0f, s1 = (_Float16)1.0f;
    if (OUTER) { off = ((c & 3) << 5) + hi * 8; s0 = scl0[c >> 2]; s1 = scl1[c >> 2]; }
    else       { off = (c << 5) + hi * 8; }
    v16h a0 = make_a<OUTER>(src0, off, s0);
    v16h a1 = make_a<OUTER>(src1, off, s1);
    int blk = (c << 3) + t0;
    // prefetch exactly the two 1KB blocks this wave loads 4 chunks from now,
    // per-lane spread, near-scope (pull into WGP$)
    __builtin_prefetch((const void*)(wf + (((size_t)blk + 32) << 9) + (lane << 4)), 0, 3);
    __builtin_prefetch((const void*)(wg + (((size_t)blk + 32) << 9) + (lane << 4)), 0, 3);
    v16h bF = load_b(wf, blk, lane);
    v16h bG = load_b(wg, blk, lane);
    f0 = WMMA_F16(a0, bF, f0);
    f1 = WMMA_F16(a1, bF, f1);
    g0 = WMMA_F16(a0, bG, g0);
    g1 = WMMA_F16(a1, bG, g1);
  }
}

// Post-GEMM: add biases, write f16 activations to LDS, fold FC contribution
// into fcacc via ds_add_f32. C layout: VGPR r, lane L -> m = r + 8*(L>=16),
// n = L&15 within the 16x16 tile. f0/g0 = rows 0..15, f1/g1 = rows 16..31.
__device__ __forceinline__ void store_layer(
    const v8f& f0, const v8f& f1, const v8f& g0, const v8f& g1,
    const float* __restrict__ bf, const float* __restrict__ bg,
    const float* __restrict__ wfc, int fcF, int fcG,
    _Float16* __restrict__ dst1, _Float16* __restrict__ dstOb, int obOff,
    float* __restrict__ fcacc, int n0w, int hi, int ln)
{
  int n = n0w + ln;
  float bfv = bf[n], bgv = bg[n];
  float wFv = wfc[fcF + n], wGv = wfc[fcG + n];
#pragma unroll
  for (int rt = 0; rt < 2; ++rt) {
    const v8f& vf = rt ? f1 : f0;
    const v8f& vg = rt ? g1 : g0;
#pragma unroll
    for (int r = 0; r < 8; ++r) {
      int m = rt * 16 + r + hi * 8;
      float av = vf[r] + bfv;
      float bv = vg[r] + bgv;
      if (dst1)  dst1[m * OT_S + n]          = (_Float16)av;
      if (dstOb) dstOb[m * OB_S + obOff + n] = (_Float16)av;
      atomicAdd(&fcacc[m], av * wFv + bv * wGv);   // ds_add_f32
    }
  }
}

// ---------------------------------------------------------------------------
// Fused PolyNet: layer1 + 2 poly layers + FC, one workgroup per 32 rows.
// 8 wave32s: wave w -> n-tile w (cols w*16..w*16+16), rows 0..31 (2 row-tiles).
// ---------------------------------------------------------------------------
__global__ __launch_bounds__(256) void polynet_main(
    const float* __restrict__ x, const float* __restrict__ bias0p,
    const float* __restrict__ bf1, const float* __restrict__ bg1,
    const float* __restrict__ bf2, const float* __restrict__ bg2,
    const float* __restrict__ bf3, const float* __restrict__ bg3,
    const float* __restrict__ wfc, const float* __restrict__ bfcp,
    const _Float16* __restrict__ pWf1, const _Float16* __restrict__ pWg1,
    const _Float16* __restrict__ pWf2, const _Float16* __restrict__ pWg2,
    const _Float16* __restrict__ pWf3, const _Float16* __restrict__ pWg3,
    float* __restrict__ out)
{
  __shared__ __align__(16) _Float16 zb [kBM * ZB_S];   // [bias | x | 0-pad]
  __shared__ __align__(16) _Float16 o1h[kBM * OT_S];   // ot1 (degree-1, fixed)
  __shared__ __align__(16) _Float16 oth[kBM * OT_S];   // xf2 (current ot)
  __shared__ __align__(16) _Float16 obh[kBM * OB_S];   // [bias | ot1 | xf2 | 0]
  __shared__ float fcacc[kBM];

  const int tid  = threadIdx.x;
  const int lane = tid & 31;
  const int wave = tid >> 5;
  const int hi = lane >> 4, ln = lane & 15;
  const int t0 = wave, n0w = wave * 16;
  const int row0 = blockIdx.x * kBM;
  const float b0 = bias0p[0];

  // ---- init LDS ----
  for (int i = tid; i < kBM * ZB_S; i += 256) zb[i]  = (_Float16)0.0f;
  for (int i = tid; i < kBM * OB_S; i += 256) obh[i] = (_Float16)0.0f;
  if (tid < kBM) fcacc[tid] = b0 * wfc[0] + bfcp[0];
  __syncthreads();
  for (int i = tid; i < kBM * kDIn; i += 256) {
    int m = i >> 6, k = i & 63;
    zb[m * ZB_S + 1 + k] = (_Float16)x[(size_t)(row0 + m) * kDIn + k];
  }
  if (tid < kBM) { zb[tid * ZB_S] = (_Float16)b0; obh[tid * OB_S] = (_Float16)b0; }
  __syncthreads();

  const int m0 = ln, m1 = 16 + ln;
  const _Float16* zbR0 = zb  + m0 * ZB_S;
  const _Float16* zbR1 = zb  + m1 * ZB_S;
  const _Float16* o1R0 = o1h + m0 * OT_S;
  const _Float16* o1R1 = o1h + m1 * OT_S;
  const _Float16* otR0 = oth + m0 * OT_S;
  const _Float16* otR1 = oth + m1 * OT_S;
  const _Float16* obR0 = obh + m0 * OB_S;
  const _Float16* obR1 = obh + m1 * OB_S;

  v8f f0, f1, g0, g1;

  // ---- layer 1: z @ {Wf1,Wg1} ----
  f0 = vzero(); f1 = vzero(); g0 = vzero(); g1 = vzero();
  gemm_seg<false>(zbR0, zbR1, o1R0, o1R1, pWf1, pWg1, 3, t0, lane, hi,
                  f0, f1, g0, g1);
  store_layer(f0, f1, g0, g1, bf1, bg1, wfc, 1, 385,
              o1h, obh, 1, fcacc, n0w, hi, ln);
  __syncthreads();

  // ---- layer 2: [o | ot1 (x) ot1] @ {Wf2,Wg2} ----
  f0 = vzero(); f1 = vzero(); g0 = vzero(); g1 = vzero();
  gemm_seg<false>(obR0, obR1, o1R0, o1R1, pWf2, pWg2, 5, t0, lane, hi,
                  f0, f1, g0, g1);
  gemm_seg<true>(o1R0, o1R1, o1R0, o1R1,
                 pWf2 + (size_t)L2O_BLKS * HPB, pWg2 + (size_t)L2O_BLKS * HPB,
                 512, t0, lane, hi, f0, f1, g0, g1);
  __syncthreads();  // obh[129..160) read above, written below
  store_layer(f0, f1, g0, g1, bf2, bg2, wfc, 129, 513,
              oth, obh, 129, fcacc, n0w, hi, ln);
  __syncthreads();

  // ---- layer 3: [o | ot1 (x) xf2] @ {Wf3,Wg3} ----
  f0 = vzero(); f1 = vzero(); g0 = vzero(); g1 = vzero();
  gemm_seg<false>(obR0, obR1, o1R0, o1R1, pWf3, pWg3, 9, t0, lane, hi,
                  f0, f1, g0, g1);
  gemm_seg<true>(otR0, otR1, o1R0, o1R1,
                 pWf3 + (size_t)L3O_BLKS * HPB, pWg3 + (size_t)L3O_BLKS * HPB,
                 512, t0, lane, hi, f0, f1, g0, g1);
  store_layer(f0, f1, g0, g1, bf3, bg3, wfc, 257, 641,
              nullptr, nullptr, 0, fcacc, n0w, hi, ln);
  __syncthreads();

  if (tid < kBM) out[row0 + tid] = fcacc[tid];
}

extern "C" void kernel_launch(void* const* d_in, const int* in_sizes, int n_in,
                              void* d_out, int out_size, void* d_ws, size_t ws_size,
                              hipStream_t stream)
{
  (void)in_sizes; (void)n_in; (void)out_size;
  const float* x   = (const float*)d_in[0];
  const float* b0  = (const float*)d_in[1];
  const float* Wf1 = (const float*)d_in[2];
  const float* bf1 = (const float*)d_in[3];
  const float* Wg1 = (const float*)d_in[4];
  const float* bg1 = (const float*)d_in[5];
  const float* Wf2 = (const float*)d_in[6];
  const float* bf2 = (const float*)d_in[7];
  const float* Wg2 = (const float*)d_in[8];
  const float* bg2 = (const float*)d_in[9];
  const float* Wf3 = (const float*)d_in[10];
  const float* bf3 = (const float*)d_in[11];
  const float* Wg3 = (const float*)d_in[12];
  const float* bg3 = (const float*)d_in[13];
  const float* Wfc = (const float*)d_in[14];
  const float* bfc = (const float*)d_in[15];
  float* out = (float*)d_out;
  _Float16* ws = (_Float16*)d_ws;
  if (ws_size < kWsHalves * sizeof(_Float16)) return;  // need ~16.3 MiB scratch

  auto pack = [&](const float* W, size_t off, int rowoff, int kvalid, int nblocks) {
    int total = nblocks << 9;
    pack_w_kernel<<<(total + 255) / 256, 256, 0, stream>>>(W, ws + off, rowoff,
                                                           kvalid, nblocks);
  };
  pack(Wf1, oWf1, 0, 65, L1_BLKS);
  pack(Wg1, oWg1, 0, 65, L1_BLKS);
  pack(Wf2, oWf2, 0, 129, L2O_BLKS);
  pack(Wf2, oWf2 + (size_t)L2O_BLKS * HPB, 129, 16384, OUT_BLKS);
  pack(Wg2, oWg2, 0, 129, L2O_BLKS);
  pack(Wg2, oWg2 + (size_t)L2O_BLKS * HPB, 129, 16384, OUT_BLKS);
  pack(Wf3, oWf3, 0, 257, L3O_BLKS);
  pack(Wf3, oWf3 + (size_t)L3O_BLKS * HPB, 257, 16384, OUT_BLKS);
  pack(Wg3, oWg3, 0, 257, L3O_BLKS);
  pack(Wg3, oWg3 + (size_t)L3O_BLKS * HPB, 257, 16384, OUT_BLKS);

  polynet_main<<<kB / kBM, 256, 0, stream>>>(
      x, b0, bf1, bg1, bf2, bg2, bf3, bg3, Wfc, bfc,
      ws + oWf1, ws + oWg1, ws + oWf2, ws + oWg2, ws + oWf3, ws + oWg3, out);
}
